// SameMovieHeadLoss_21964462752212
// MI455X (gfx1250) — matrix operand order repaired and checked by reading, not compiled, not run on hardware
//
#include <hip/hip_runtime.h>
#include <hip/hip_bf16.h>
#include <math.h>

#define N_ROWS   8192
#define HIDDEN   1024
#define PROJ     128
#define HALF_N   4096
#define INV_T    (1.0f / 0.07f)
#define LN_EPS   1e-12f

typedef __attribute__((ext_vector_type(16))) __bf16 v16bf;
typedef __attribute__((ext_vector_type(8)))  __bf16 v8bf;
typedef __attribute__((ext_vector_type(8)))  float  v8f;

static __device__ __forceinline__ v8f wmma_bf16(v16bf a, v16bf b, v8f c) {
  // D = A(16x32 bf16) x B(32x16 bf16) + C(16x16 f32)
  return __builtin_amdgcn_wmma_f32_16x16x32_bf16(
      /*neg_a=*/false, a, /*neg_b=*/false, b,
      /*c_mod=*/(short)0, c, /*reuse_a=*/false, /*reuse_b=*/false);
}

// A operand (16x32, 16-bit): lane l holds row (l&15); K contiguous:
//   k = kBase + 16*(l>>4) + [0..16).
static __device__ __forceinline__ v16bf load_a_from_f32(const float* p0) {
  float4 f0 = *(const float4*)(p0);
  float4 f1 = *(const float4*)(p0 + 4);
  float4 f2 = *(const float4*)(p0 + 8);
  float4 f3 = *(const float4*)(p0 + 12);
  v16bf a;
  a[0]=(__bf16)f0.x;  a[1]=(__bf16)f0.y;  a[2]=(__bf16)f0.z;  a[3]=(__bf16)f0.w;
  a[4]=(__bf16)f1.x;  a[5]=(__bf16)f1.y;  a[6]=(__bf16)f1.z;  a[7]=(__bf16)f1.w;
  a[8]=(__bf16)f2.x;  a[9]=(__bf16)f2.y;  a[10]=(__bf16)f2.z; a[11]=(__bf16)f2.w;
  a[12]=(__bf16)f3.x; a[13]=(__bf16)f3.y; a[14]=(__bf16)f3.z; a[15]=(__bf16)f3.w;
  return a;
}

static __device__ __forceinline__ v16bf load_a_from_bf16(const __bf16* p0) {
  v8bf c0 = *(const v8bf*)(p0);
  v8bf c1 = *(const v8bf*)(p0 + 8);
  v16bf a;
#pragma unroll
  for (int i = 0; i < 8; ++i) { a[i] = c0[i]; a[i + 8] = c1[i]; }
  return a;
}

// ---------------------------------------------------------------------------
// Prep: transpose + f32->bf16.  wT[n*K + k] = w[k*Ncols + n]
// ---------------------------------------------------------------------------
__global__ __launch_bounds__(256)
void k_transpose_bf16(const float* __restrict__ w, __bf16* __restrict__ wT,
                      int K, int Ncols) {
  int idx = blockIdx.x * 256 + threadIdx.x;
  int n = idx / K;
  int k = idx - n * K;
  if (n < Ncols) wT[idx] = (__bf16)w[(size_t)k * Ncols + n];
}

// ---------------------------------------------------------------------------
// K1: h_ln = layernorm(gelu(x @ W1 + b1)) * gamma + beta   (bf16 out)
// block = 8 waves, 32 rows (2 row-tiles share each B load);
// wave w owns columns [w*128, w*128+128)
// ---------------------------------------------------------------------------
__global__ __launch_bounds__(256)
void k_dense_gelu_ln(const float* __restrict__ x, const __bf16* __restrict__ wbT,
                     const float* __restrict__ bias, const float* __restrict__ gamma,
                     const float* __restrict__ beta, __bf16* __restrict__ hln) {
  __shared__ float ldsS[8][32];
  __shared__ float ldsQ[8][32];
  __shared__ float ldsMu[32];
  __shared__ float ldsR[32];

  const int wave = threadIdx.x >> 5;
  const int lane = threadIdx.x & 31;
  const int hf   = lane >> 4;
  const int l16  = lane & 15;
  const int rowBase = blockIdx.x * 32;

  const float* xrow0 = x + (size_t)(rowBase + l16) * HIDDEN;
  const float* xrow1 = x + (size_t)(rowBase + 16 + l16) * HIDDEN;

  v8f c[2][8];
#pragma unroll
  for (int rt = 0; rt < 2; ++rt)
#pragma unroll
    for (int t = 0; t < 8; ++t)
#pragma unroll
      for (int i = 0; i < 8; ++i) c[rt][t][i] = 0.0f;

  for (int k0 = 0; k0 < HIDDEN; k0 += 32) {
    v16bf a0 = load_a_from_f32(xrow0 + k0 + 16 * hf);
    v16bf a1 = load_a_from_f32(xrow1 + k0 + 16 * hf);
#pragma unroll
    for (int t = 0; t < 8; ++t) {
      int coln = wave * 128 + t * 16 + l16;  // B column; lanes 16-31: K+16
      v16bf b = *(const v16bf*)(wbT + (size_t)coln * HIDDEN + k0 + 16 * hf);
      c[0][t] = wmma_bf16(a0, b, c[0][t]);
      c[1][t] = wmma_bf16(a1, b, c[1][t]);
    }
  }

  // bias + exact (erf) gelu
#pragma unroll
  for (int t = 0; t < 8; ++t) {
    int coln = wave * 128 + t * 16 + l16;
    float bb = bias[coln];
#pragma unroll
    for (int rt = 0; rt < 2; ++rt)
#pragma unroll
      for (int v = 0; v < 8; ++v) {
        float h = c[rt][t][v] + bb;
        c[rt][t][v] = 0.5f * h * (1.0f + erff(h * 0.70710678118654752f));
      }
  }

  // layernorm statistics: C layout row = rt*16 + v + 8*hf, col = lane%16
#pragma unroll
  for (int rt = 0; rt < 2; ++rt)
#pragma unroll
    for (int v = 0; v < 8; ++v) {
      float s = 0.0f, q = 0.0f;
#pragma unroll
      for (int t = 0; t < 8; ++t) { float h = c[rt][t][v]; s += h; q += h * h; }
#pragma unroll
      for (int mask = 1; mask <= 8; mask <<= 1) {
        s += __shfl_xor(s, mask, 32);
        q += __shfl_xor(q, mask, 32);
      }
      if (l16 == 0) {
        ldsS[wave][rt * 16 + v + 8 * hf] = s;
        ldsQ[wave][rt * 16 + v + 8 * hf] = q;
      }
    }
  __syncthreads();
  if (threadIdx.x < 32) {
    float S = 0.0f, Q = 0.0f;
    for (int w = 0; w < 8; ++w) { S += ldsS[w][threadIdx.x]; Q += ldsQ[w][threadIdx.x]; }
    float mu  = S * (1.0f / HIDDEN);
    float var = Q * (1.0f / HIDDEN) - mu * mu;
    ldsMu[threadIdx.x] = mu;
    ldsR[threadIdx.x]  = rsqrtf(var + LN_EPS);
  }
  __syncthreads();

#pragma unroll
  for (int t = 0; t < 8; ++t) {
    int coln = wave * 128 + t * 16 + l16;
    float g = gamma[coln], be = beta[coln];
#pragma unroll
    for (int rt = 0; rt < 2; ++rt)
#pragma unroll
      for (int v = 0; v < 8; ++v) {
        int rl = rt * 16 + v + 8 * hf;
        float y = (c[rt][t][v] - ldsMu[rl]) * ldsR[rl] * g + be;
        hln[(size_t)(rowBase + rl) * HIDDEN + coln] = (__bf16)y;
      }
  }
}

// ---------------------------------------------------------------------------
// K2: zb = bf16(h_ln @ W2 + b2)   [8192 x 128]
// block = 8 waves, 32 rows; wave w owns 16 cols; B load shared by 2 row-tiles
// ---------------------------------------------------------------------------
__global__ __launch_bounds__(256)
void k_decoder(const __bf16* __restrict__ hln, const __bf16* __restrict__ dwT,
               const float* __restrict__ bias, __bf16* __restrict__ zb) {
  const int wave = threadIdx.x >> 5;
  const int lane = threadIdx.x & 31;
  const int hf   = lane >> 4;
  const int l16  = lane & 15;
  const int rowBase = blockIdx.x * 32;
  const int coln = wave * 16 + l16;

  const __bf16* arow0 = hln + (size_t)(rowBase + l16) * HIDDEN;
  const __bf16* arow1 = hln + (size_t)(rowBase + 16 + l16) * HIDDEN;

  v8f c0, c1;
#pragma unroll
  for (int i = 0; i < 8; ++i) { c0[i] = 0.0f; c1[i] = 0.0f; }

  for (int k0 = 0; k0 < HIDDEN; k0 += 32) {
    v16bf a0 = load_a_from_bf16(arow0 + k0 + 16 * hf);
    v16bf a1 = load_a_from_bf16(arow1 + k0 + 16 * hf);
    v16bf b  = *(const v16bf*)(dwT + (size_t)coln * HIDDEN + k0 + 16 * hf);
    c0 = wmma_bf16(a0, b, c0);
    c1 = wmma_bf16(a1, b, c1);
  }

  float bb = bias[coln];
#pragma unroll
  for (int v = 0; v < 8; ++v) {
    zb[(size_t)(rowBase + v + 8 * hf) * PROJ + coln]      = (__bf16)(c0[v] + bb);
    zb[(size_t)(rowBase + 16 + v + 8 * hf) * PROJ + coln] = (__bf16)(c1[v] + bb);
  }
}

// ---------------------------------------------------------------------------
// K3: streaming sims = z @ z.T fused with online logsumexp.
// Per row i: loss_i = logsumexp_{j!=i}(sims[i,j]/T) - sims[i, i^4096]/T
// block = 8 waves, 32 rows; wave w streams columns [w*1024, w*1024+1024);
// each B load feeds 2 WMMAs (two row-tiles).
// ---------------------------------------------------------------------------
__global__ __launch_bounds__(256)
void k_sims_loss(const __bf16* __restrict__ zb, float* __restrict__ rowloss) {
  __shared__ float lm[8][32];
  __shared__ float lsum[8][32];
  __shared__ float lpos[8][32];

  const int wave = threadIdx.x >> 5;
  const int lane = threadIdx.x & 31;
  const int hf   = lane >> 4;
  const int l16  = lane & 15;
  const int rowBase = blockIdx.x * 32;
  const float NEG_INF = -__builtin_inff();

  // preload A: rows [rowBase, rowBase+32), K = 0..127 -> 2x4 register tiles
  v16bf a[2][4];
#pragma unroll
  for (int rt = 0; rt < 2; ++rt) {
    const __bf16* arow = zb + (size_t)(rowBase + rt * 16 + l16) * PROJ;
#pragma unroll
    for (int kk = 0; kk < 4; ++kk)
      a[rt][kk] = load_a_from_bf16(arow + kk * 32 + 16 * hf);
  }

  float m[2][8], s[2][8], lp[2][8];
#pragma unroll
  for (int rt = 0; rt < 2; ++rt)
#pragma unroll
    for (int v = 0; v < 8; ++v) { m[rt][v] = NEG_INF; s[rt][v] = 0.0f; lp[rt][v] = 0.0f; }

  for (int ct = 0; ct < 64; ++ct) {
    const int colBase = wave * 1024 + ct * 16;
    const int col = colBase + l16;
    // B column j == row j of zb (sims = z z^T); lanes 16-31 take K+16
    const __bf16* brow = zb + (size_t)col * PROJ + 16 * hf;

    v8f c0, c1;
#pragma unroll
    for (int i = 0; i < 8; ++i) { c0[i] = 0.0f; c1[i] = 0.0f; }
#pragma unroll
    for (int kk = 0; kk < 4; ++kk) {
      v16bf b = *(const v16bf*)(brow + kk * 32);
      c0 = wmma_bf16(a[0][kk], b, c0);
      c1 = wmma_bf16(a[1][kk], b, c1);
    }

#pragma unroll
    for (int rt = 0; rt < 2; ++rt) {
#pragma unroll
      for (int v = 0; v < 8; ++v) {
        int row = rowBase + rt * 16 + v + 8 * hf;
        float xv = (rt == 0 ? c0[v] : c1[v]) * INV_T;
        if (col == (row ^ HALF_N)) lp[rt][v] += xv;  // positive logit
        if (col != row) {                            // exclude self-similarity
          // branchy online lse: common case costs a single exp
          if (xv <= m[rt][v]) {
            s[rt][v] += __expf(xv - m[rt][v]);
          } else {
            s[rt][v] = s[rt][v] * __expf(m[rt][v] - xv) + 1.0f;
            m[rt][v] = xv;
          }
        }
      }
    }
  }

  // reduce (m,s,lp) across the 16 lanes of each half-wave
#pragma unroll
  for (int rt = 0; rt < 2; ++rt)
#pragma unroll
    for (int v = 0; v < 8; ++v) {
#pragma unroll
      for (int mask = 1; mask <= 8; mask <<= 1) {
        float om = __shfl_xor(m[rt][v], mask, 32);
        float os = __shfl_xor(s[rt][v], mask, 32);
        float nm = fmaxf(m[rt][v], om);
        s[rt][v] = s[rt][v] * __expf(m[rt][v] - nm) + os * __expf(om - nm);
        m[rt][v] = nm;
        lp[rt][v] += __shfl_xor(lp[rt][v], mask, 32);
      }
      if (l16 == 0) {
        lm[wave][rt * 16 + v + 8 * hf]   = m[rt][v];
        lsum[wave][rt * 16 + v + 8 * hf] = s[rt][v];
        lpos[wave][rt * 16 + v + 8 * hf] = lp[rt][v];
      }
    }
  __syncthreads();

  if (threadIdx.x < 32) {
    int r = threadIdx.x;
    float M = -__builtin_inff();
    for (int w = 0; w < 8; ++w) M = fmaxf(M, lm[w][r]);
    float S = 0.0f, LP = 0.0f;
    for (int w = 0; w < 8; ++w) {
      S  += lsum[w][r] * __expf(lm[w][r] - M);
      LP += lpos[w][r];
    }
    rowloss[rowBase + r] = M + logf(S) - LP;
  }
}

// ---------------------------------------------------------------------------
// K4: deterministic mean over 8192 row losses
// ---------------------------------------------------------------------------
__global__ __launch_bounds__(256)
void k_mean(const float* __restrict__ rowloss, float* __restrict__ out) {
  __shared__ float lds[256];
  float acc = 0.0f;
  for (int i = threadIdx.x; i < N_ROWS; i += 256) acc += rowloss[i];
  lds[threadIdx.x] = acc;
  __syncthreads();
  for (int st = 128; st > 0; st >>= 1) {
    if (threadIdx.x < st) lds[threadIdx.x] += lds[threadIdx.x + st];
    __syncthreads();
  }
  if (threadIdx.x == 0) out[0] = lds[0] * (1.0f / N_ROWS);
}

// ---------------------------------------------------------------------------
extern "C" void kernel_launch(void* const* d_in, const int* in_sizes, int n_in,
                              void* d_out, int out_size, void* d_ws, size_t ws_size,
                              hipStream_t stream) {
  (void)in_sizes; (void)n_in; (void)out_size; (void)ws_size;

  const float* x        = (const float*)d_in[0];
  const float* dense_w  = (const float*)d_in[1];
  const float* dense_b  = (const float*)d_in[2];
  const float* ln_gamma = (const float*)d_in[3];
  const float* ln_beta  = (const float*)d_in[4];
  const float* dec_w    = (const float*)d_in[5];
  const float* dec_b    = (const float*)d_in[6];

  char* ws = (char*)d_ws;
  size_t off = 0;
  __bf16* wbT = (__bf16*)(ws + off); off += (size_t)HIDDEN * HIDDEN * 2;  //  2 MB
  __bf16* dwT = (__bf16*)(ws + off); off += (size_t)PROJ * HIDDEN * 2;    // 256 KB
  __bf16* hln = (__bf16*)(ws + off); off += (size_t)N_ROWS * HIDDEN * 2;  // 16 MB
  __bf16* zb  = (__bf16*)(ws + off); off += (size_t)N_ROWS * PROJ * 2;    //  2 MB
  float* rowloss = (float*)(ws + off);                                    // 32 KB

  // weight transpose/convert (W1 -> [n][k], W2 -> [n][k])
  k_transpose_bf16<<<(HIDDEN * HIDDEN) / 256, 256, 0, stream>>>(dense_w, wbT, HIDDEN, HIDDEN);
  k_transpose_bf16<<<(PROJ * HIDDEN) / 256, 256, 0, stream>>>(dec_w, dwT, HIDDEN, PROJ);

  // fused dense + gelu + layernorm (32 rows / block)
  k_dense_gelu_ln<<<N_ROWS / 32, 256, 0, stream>>>(x, wbT, dense_b, ln_gamma, ln_beta, hln);

  // decoder projection (32 rows / block)
  k_decoder<<<N_ROWS / 32, 256, 0, stream>>>(hln, dwT, dec_b, zb);

  // fused z z^T + online logsumexp loss (sims never materialized)
  k_sims_loss<<<N_ROWS / 32, 256, 0, stream>>>(zb, rowloss);

  // deterministic mean
  k_mean<<<1, 256, 0, stream>>>(rowloss, (float*)d_out);
}
